// ScaledDotProductAttention_69965017252226
// MI455X (gfx1250) — matrix operand order repaired and checked by reading, not compiled, not run on hardware
//
#include <hip/hip_runtime.h>
#include <hip/hip_bf16.h>

// MI455X / gfx1250 fused masked attention.
// out tuple = (output [16,2048,64] f32, attn [16,2048,2048] f32), concatenated in d_out.

typedef __attribute__((ext_vector_type(16))) __bf16 v16bf;
typedef __attribute__((ext_vector_type(8)))  float  v8f;

#define BATCH 16
#define SEQ   2048
#define DHEAD 64

static __device__ __forceinline__ __bf16 tobf(float x) { return (__bf16)x; }

// ---------------------------------------------------------------------------
// Kernel 0a: pack V (f32 [B, L, 64]) into WMMA B-operand bf16 layout for P*V.
// vb[((b*64 + pp)*4 + dt)*512 + lane*16 + e]
//   = V[b][pp*32 + 16*(lane/16) + e][dt*16 + (lane%16)]
// ---------------------------------------------------------------------------
__global__ __launch_bounds__(256) void pack_v_kernel(const float* __restrict__ v,
                                                     __bf16* __restrict__ vb) {
  int gid  = blockIdx.x * 256 + threadIdx.x;   // 0 .. 16*64*4*32-1
  int lane = gid & 31;
  int dt   = (gid >> 5) & 3;
  int pp   = (gid >> 7) & 63;
  int b    = gid >> 13;
  int h    = lane >> 4;
  int c    = lane & 15;

  __bf16* dst = vb + (size_t)gid * 16;
  const float* src = v + ((size_t)b * SEQ + pp * 32 + h * 16) * DHEAD + dt * 16 + c;
#pragma unroll
  for (int e = 0; e < 16; ++e) dst[e] = tobf(src[(size_t)e * DHEAD]);
}

// ---------------------------------------------------------------------------
// Kernel 0b: pack K (f32 [B, L, 64]) into WMMA B-operand bf16 layout for Q*K^T.
// kb[((b*128 + kt)*2 + half)*512 + lane*16 + e]
//   = K[b][kt*16 + (lane%16)][32*half + 16*(lane/16) + e]
// ---------------------------------------------------------------------------
__global__ __launch_bounds__(256) void pack_k_kernel(const float* __restrict__ k,
                                                     __bf16* __restrict__ kb) {
  int gid  = blockIdx.x * 256 + threadIdx.x;   // 0 .. 16*128*2*32-1
  int lane = gid & 31;
  int half = (gid >> 5) & 1;
  int kt   = (gid >> 6) & 127;
  int b    = gid >> 13;
  int h    = lane >> 4;
  int c    = lane & 15;

  __bf16* dst = kb + (size_t)gid * 16;
  const float* src = k + ((size_t)b * SEQ + kt * 16 + c) * DHEAD + half * 32 + h * 16;
#pragma unroll
  for (int e = 0; e < 16; ++e) dst[e] = tobf(src[e]);
}

// ---------------------------------------------------------------------------
// Main kernel: one block (8 wave32) per (batch b, 16-row query tile qt).
// ---------------------------------------------------------------------------
__global__ __launch_bounds__(256) void attn_kernel(const float* __restrict__ q,
                                                   const __bf16* __restrict__ kb,
                                                   const int*   __restrict__ mask,
                                                   const __bf16* __restrict__ vb,
                                                   float* __restrict__ out,
                                                   float* __restrict__ attn) {
  const int b     = blockIdx.y;
  const int qt    = blockIdx.x;
  const int qbase = qt * 16;
  const int tid   = threadIdx.x;
  const int w     = tid >> 5;     // wave id 0..7
  const int l     = tid & 31;     // lane
  const int c16   = l & 15;
  const int h     = l >> 4;       // lane half

  __shared__ float lds_e[8][16][33];   // per-wave C->A relayout buffer
  __shared__ float lds_max[8][16];
  __shared__ float lds_sum[8][16];
  __shared__ float lds_fmax[16];
  __shared__ float lds_den[16];
  __shared__ float lds_o[16][64];

  // zero O accumulator (covered by the phase-1 end barrier)
  for (int i = tid; i < 16 * 64; i += 256) (&lds_o[0][0])[i] = 0.0f;

  // ---- Build Q A-operands once per wave (A: lane holds row M = lane%16) ----
  // A element e -> contraction k = 16*h + e (contiguous order, matches packed K).
  const float* qrow = q + ((size_t)b * SEQ + qbase + c16) * DHEAD;
  v16bf a0, a1;
#pragma unroll
  for (int e = 0; e < 16; ++e) {
    int d0 = (h << 4) + e;
    a0[e] = tobf(qrow[d0]);
    a1[e] = tobf(qrow[d0 + 32]);
  }

  // ---- Phase 1: scores for 16 k-tiles per wave, kept in VGPRs -------------
  float tr[16][8];     // masked clipped scores, C-layout (row m = r + 8h, col n = c16)
  float rmax[8];
#pragma unroll
  for (int r = 0; r < 8; ++r) rmax[r] = -1e30f;

  const size_t rowbase = (size_t)b * SEQ + qbase;
  const size_t mlane0  = (rowbase + (h << 3)) * (size_t)SEQ + c16;  // r=0 mask addr base

#pragma unroll
  for (int i = 0; i < 16; ++i) {
    const int kt = w * 16 + i;

    // pre-packed bf16 K B-operands: 2 x 32B per lane, L2-resident
    const __bf16* kbp = kb + (((size_t)b * 128 + kt) * 2) * 512 + l * 16;
    v16bf b0 = *(const v16bf*)(kbp);
    v16bf b1 = *(const v16bf*)(kbp + 512);

    // batch the 8 mask loads (the HBM-heavy stream) so they clause and
    // overlap the WMMA latency instead of load->wait->use chains
    int mv[8];
#pragma unroll
    for (int r = 0; r < 8; ++r)
      mv[r] = mask[mlane0 + (size_t)r * SEQ + (size_t)kt * 16];

    // prefetch next tile's mask rows into near caches
    if (i < 15) __builtin_prefetch(mask + mlane0 + (size_t)(kt + 1) * 16, 0, 3);

    v8f c = {};
    c = __builtin_amdgcn_wmma_f32_16x16x32_bf16(false, a0, false, b0, (short)0, c, false, false);
    c = __builtin_amdgcn_wmma_f32_16x16x32_bf16(false, a1, false, b1, (short)0, c, false, false);

#pragma unroll
    for (int r = 0; r < 8; ++r) {
      float s = c[r] * 0.125f;                       // 1/TEMPERATURE
      s = fminf(fmaxf(s, -15.0f), 15.0f);            // clip
      // branch-free mask fold: mf=1 -> s ; mf=0 -> -1e30 sentinel (exp -> 0)
      float mf = (float)mv[r];
      float sm = s * mf;                             // x * mask
      rmax[r]  = fmaxf(rmax[r], sm);                 // max over x*mask
      tr[i][r] = sm + (mf * 1e30f - 1e30f);
    }
  }

  // cross-lane (within 16-lane half) and cross-wave row-max reduction
#pragma unroll
  for (int r = 0; r < 8; ++r) {
    float v = rmax[r];
    v = fmaxf(v, __shfl_xor(v, 1, 32));
    v = fmaxf(v, __shfl_xor(v, 2, 32));
    v = fmaxf(v, __shfl_xor(v, 4, 32));
    v = fmaxf(v, __shfl_xor(v, 8, 32));
    rmax[r] = v;
  }
  if (c16 == 0) {
#pragma unroll
    for (int r = 0; r < 8; ++r) lds_max[w][r + (h << 3)] = rmax[r];
  }
  __syncthreads();
  if (tid < 16) {
    float m = lds_max[0][tid];
    for (int ww = 1; ww < 8; ++ww) m = fmaxf(m, lds_max[ww][tid]);
    lds_fmax[tid] = m;
  }
  __syncthreads();

  // ---- Phase 2: e = exp(t - max); O += e * V (bf16 WMMA); row sums --------
  const float fmax_m = lds_fmax[c16];   // A-layout row of this lane
  v8f o0 = {}, o1 = {}, o2 = {}, o3 = {};
  float rsum = 0.0f;

#pragma unroll
  for (int j = 0; j < 8; ++j) {
    // stage two C-layout tiles (k-rel 0..31) into wave-local LDS
#pragma unroll
    for (int half = 0; half < 2; ++half) {
#pragma unroll
      for (int r = 0; r < 8; ++r)
        lds_e[w][r + (h << 3)][half * 16 + c16] = tr[2 * j + half][r];
    }
    // gather in A layout, exponentiate, accumulate row sum
    v16bf ae;
#pragma unroll
    for (int e = 0; e < 16; ++e) {
      int krel = (h << 4) + e;
      float ev = __expf(lds_e[w][c16][krel] - fmax_m);
      rsum += ev;
      ae[e] = tobf(ev);
    }
    // V B-operands, pre-packed bf16, 32B per lane, L2-resident
    const int pp = w * 8 + j;
    const __bf16* vbp = vb + (((size_t)b * 64 + pp) * 4) * 512 + l * 16;
    v16bf bv0 = *(const v16bf*)(vbp);
    v16bf bv1 = *(const v16bf*)(vbp + 512);
    v16bf bv2 = *(const v16bf*)(vbp + 1024);
    v16bf bv3 = *(const v16bf*)(vbp + 1536);
    o0 = __builtin_amdgcn_wmma_f32_16x16x32_bf16(false, ae, false, bv0, (short)0, o0, false, false);
    o1 = __builtin_amdgcn_wmma_f32_16x16x32_bf16(false, ae, false, bv1, (short)0, o1, false, false);
    o2 = __builtin_amdgcn_wmma_f32_16x16x32_bf16(false, ae, false, bv2, (short)0, o2, false, false);
    o3 = __builtin_amdgcn_wmma_f32_16x16x32_bf16(false, ae, false, bv3, (short)0, o3, false, false);
  }

  // row-sum reduction: halves hold disjoint k subsets of the same row
  rsum += __shfl_xor(rsum, 16, 32);
  if (l < 16) lds_sum[w][l] = rsum;
  __syncthreads();
  if (tid < 16) {
    float s = 0.0f;
    for (int ww = 0; ww < 8; ++ww) s += lds_sum[ww][tid];
    lds_den[tid] = 1.0f / (s + 1e-6f);
  }
  __syncthreads();

  // ---- cross-wave O reduction (ds_add_f32 atomics) ------------------------
#pragma unroll
  for (int r = 0; r < 8; ++r) {
    int m = r + (h << 3);
    atomicAdd(&lds_o[m][ 0 + c16], o0[r]);
    atomicAdd(&lds_o[m][16 + c16], o1[r]);
    atomicAdd(&lds_o[m][32 + c16], o2[r]);
    atomicAdd(&lds_o[m][48 + c16], o3[r]);
  }
  __syncthreads();

  // ---- write output (normalized), coalesced float4 ------------------------
  {
    int m  = tid >> 4;
    int d0 = (tid & 15) << 2;
    float idn = lds_den[m];
    float4 val;
    val.x = lds_o[m][d0 + 0] * idn;
    val.y = lds_o[m][d0 + 1] * idn;
    val.z = lds_o[m][d0 + 2] * idn;
    val.w = lds_o[m][d0 + 3] * idn;
    *(float4*)(out + (rowbase + m) * (size_t)DHEAD + d0) = val;
  }

  // ---- Phase 3: final attn written ONCE from registers --------------------
  float fm[8], idn[8];
#pragma unroll
  for (int r = 0; r < 8; ++r) {
    fm[r]  = lds_fmax[r + (h << 3)];
    idn[r] = lds_den[r + (h << 3)];
  }
#pragma unroll
  for (int i = 0; i < 16; ++i) {
    const size_t colbase = ((size_t)(w * 16 + i)) * 16 + c16;
#pragma unroll
    for (int r = 0; r < 8; ++r) {
      float e = __expf(tr[i][r] - fm[r]) * idn[r];
      attn[(rowbase + r + (h << 3)) * (size_t)SEQ + colbase] = e;
    }
  }
}

// ---------------------------------------------------------------------------
extern "C" void kernel_launch(void* const* d_in, const int* in_sizes, int n_in,
                              void* d_out, int out_size, void* d_ws, size_t ws_size,
                              hipStream_t stream) {
  const float* q    = (const float*)d_in[0];
  const float* k    = (const float*)d_in[1];
  const float* v    = (const float*)d_in[2];
  const int*   mask = (const int*)d_in[3];

  float* out  = (float*)d_out;
  float* attn = out + (size_t)BATCH * SEQ * DHEAD;   // tuple order: (output, attn)

  __bf16* vb  = (__bf16*)d_ws;                                 // 4 MB packed V
  __bf16* kbp = (__bf16*)((char*)d_ws + ((size_t)4 << 20));    // 4 MB packed K

  pack_v_kernel<<<512, 256, 0, stream>>>(v, vb);
  pack_k_kernel<<<512, 256, 0, stream>>>(k, kbp);

  dim3 grid(SEQ / 16, BATCH);
  attn_kernel<<<grid, 256, 0, stream>>>(q, kbp, mask, vb, out, attn);
}
